// Conv_31447750541738
// MI455X (gfx1250) — matrix operand order repaired and checked by reading, not compiled, not run on hardware
//
#include <hip/hip_runtime.h>

// Problem constants (match reference)
#define Bc   2
#define Nn   8192
#define Mm   8192
#define CIN  32
#define COUT 32
#define Ee   16
#define Kk   32
#define GAMMA 256.0f   // E*E

typedef __attribute__((ext_vector_type(16))) __bf16 v16bf;
typedef __attribute__((ext_vector_type(8)))  float  v8f;

union BF16x16 { v16bf v; unsigned short u[16]; unsigned int w[8]; uint4 q[2]; };

__device__ __forceinline__ unsigned short f2bf(float x) {
  union { __bf16 h; unsigned short u; } c;
  c.h = (__bf16)x;                       // fptrunc f32->bf16 (RNE)
  return c.u;
}

__device__ __forceinline__ unsigned int f2bf2(float lo, float hi) {
  typedef __attribute__((ext_vector_type(2))) __bf16 v2bf;
  union { v2bf v; unsigned int u; } c;
  c.v[0] = (__bf16)lo;
  c.v[1] = (__bf16)hi;
  return c.u;
}

// ---------------------------------------------------------------------------
// Kernel 0: repack coeff[i,o,e] -> CflatT[o][c] (c = i*16+e) as bf16 in ws.
// ---------------------------------------------------------------------------
__global__ void prep_cflat(const float* __restrict__ coeff,
                           unsigned short* __restrict__ cflatT) {
  int t = blockIdx.x * 256 + threadIdx.x;          // 0 .. 16383
  if (t < COUT * Ee * CIN) {
    int o = t >> 9;
    int c = t & 511;
    int i = c >> 4;
    int e = c & 15;
    cflatT[t] = f2bf(coeff[(i * COUT + o) * Ee + e]);
  }
}

// ---------------------------------------------------------------------------
// Kernel 1: brute-force kNN. One thread per output point; per-thread 32-entry
// max-heap in LDS with 64-bit keys (distBits<<32 | idx) so (dist, idx)
// lexicographic order matches jax.lax.top_k tie-breaking. Candidate stream is
// wave-uniform -> scalar-cache friendly. Float-first compare on the hot path;
// exact 64-bit key compare only on accept. Emits sorted uiv_k and idx.
// ---------------------------------------------------------------------------
__global__ __launch_bounds__(256) void knn_kernel(
    const float* __restrict__ pin,   // [B,N,2]
    const float* __restrict__ pout,  // [B,M,2]
    float* __restrict__ uiv_out,     // [B,M,K,2]
    int*   __restrict__ idx_out) {   // [B,M,K]
  __shared__ unsigned long long heap[256 * Kk];    // 64 KB, per-thread slice
  const int tid = threadIdx.x;
  const int b   = blockIdx.y;
  const int m   = blockIdx.x * 256 + tid;

  const float2* pinB = (const float2*)(pin + (size_t)b * Nn * 2);
  const float pox = pout[((size_t)b * Mm + m) * 2 + 0];
  const float poy = pout[((size_t)b * Mm + m) * 2 + 1];

  const unsigned long long INFKEY = (unsigned long long)0x7F800000u << 32;
#pragma unroll
  for (int j = 0; j < Kk; ++j) heap[j * 256 + tid] = INFKEY;
  unsigned long long root = INFKEY;
  float rootDist = __uint_as_float((unsigned)(root >> 32));

  for (int n = 0; n < Nn; ++n) {
    float2 p = pinB[n];                            // uniform across wave
    float dx = pox - p.x;
    float dy = poy - p.y;
    float d2 = fmaf(dx, dx, dy * dy);
    if (d2 <= rootDist) {                          // cheap f32 early-out
      unsigned long long key =
          ((unsigned long long)__float_as_uint(d2) << 32) | (unsigned)n;
      if (key < root) {                            // exact (dist, idx) order
        unsigned long long cur = key;
        int j = 0;
        while (true) {
          int l = 2 * j + 1;
          if (l >= Kk) break;
          unsigned long long kl = heap[l * 256 + tid];
          int r2 = l + 1;
          unsigned long long kr = (r2 < Kk) ? heap[r2 * 256 + tid] : 0ull;
          int big; unsigned long long kb;
          if (kr > kl) { big = r2; kb = kr; } else { big = l; kb = kl; }
          if (kb <= cur) break;
          heap[j * 256 + tid] = kb;
          j = big;
        }
        heap[j * 256 + tid] = cur;
        root = heap[tid];                          // slot 0
        rootDist = __uint_as_float((unsigned)(root >> 32));
      }
    }
  }

  // in-place heapsort -> ascending (dist, idx) in slots 0..31
  for (int s = Kk - 1; s > 0; --s) {
    unsigned long long mx  = heap[tid];
    unsigned long long cur = heap[s * 256 + tid];
    heap[s * 256 + tid] = mx;
    int j = 0;
    while (true) {
      int l = 2 * j + 1;
      if (l >= s) break;
      unsigned long long kl = heap[l * 256 + tid];
      int r2 = l + 1;
      unsigned long long kr = (r2 < s) ? heap[r2 * 256 + tid] : 0ull;
      int big; unsigned long long kb;
      if (kr > kl) { big = r2; kb = kr; } else { big = l; kb = kl; }
      if (kb <= cur) break;
      heap[j * 256 + tid] = kb;
      j = big;
    }
    heap[j * 256 + tid] = cur;
  }

  const size_t bm = (size_t)b * Mm + m;
  for (int k = 0; k < Kk; ++k) {
    unsigned long long key = heap[k * 256 + tid];
    unsigned nidx = (unsigned)(key & 0xffffffffu);
    float2 p = pinB[nidx];
    uiv_out[(bm * Kk + k) * 2 + 0] = pox - p.x;
    uiv_out[(bm * Kk + k) * 2 + 1] = poy - p.y;
    idx_out[bm * Kk + k] = (int)nidx;
  }
}

// ---------------------------------------------------------------------------
// Kernel 2: the einsum via two bf16 WMMA GEMMs.
//   step A (per point): G[e,i] = sum_k exp(-g*(r_k - mu_e)^2) * vals[k,i]
//       -> one v_wmma_f32_16x16x32_bf16 per CIN-half (K=32 in one op).
//   step B (16 points batched): out[p,o] = sum_c Gflat[p,c] * Cflat[c,o]
//       -> 16 k-chunks x 2 o-halves of v_wmma_f32_16x16x32_bf16.
// Gflat ordering c = i*16+e makes every fragment access b128-vectorizable.
// 2 waves per block, 16 output points per wave.
// ---------------------------------------------------------------------------
__global__ __launch_bounds__(64) void conv_kernel(
    const float* __restrict__ vin,            // [B,N,CIN]
    const float* __restrict__ bias,           // [COUT]
    const unsigned short* __restrict__ cflatT,// [COUT][512] bf16, c = i*16+e
    const float* __restrict__ uiv,            // [B,M,K,2]
    const int*   __restrict__ idx,            // [B,M,K]
    float* __restrict__ out) {                // [B,M,COUT]
  __shared__ __align__(16) float          rbuf[2][16][Kk];   //  4 KB
  __shared__ __align__(16) int            ibuf[2][16][Kk];   //  4 KB
  __shared__ __align__(16) unsigned short Gbuf[2][16][512];  // 32 KB
  __shared__ float sbias[COUT];

  const int tid    = threadIdx.x;        // 0..63
  const int w      = tid >> 5;           // wave id in block
  const int lane   = tid & 31;
  const int bmBase = blockIdx.x * 32;    // 32 points per block
  const int pBase  = bmBase + w * 16;    // first global bm index for this wave
  const int b      = pBase >> 13;        // M = 8192

  if (tid < COUT) sbias[tid] = bias[tid];

  // stage r_k (recomputed from uiv) and idx: lane = k, z = point
  for (int z = 0; z < 16; ++z) {
    size_t off = ((size_t)pBase + z) * Kk + lane;
    float ux = uiv[off * 2 + 0];
    float uy = uiv[off * 2 + 1];
    rbuf[w][z][lane] = sqrtf(fmaf(ux, ux, uy * uy));
    ibuf[w][z][lane] = idx[off];
  }
  __syncthreads();

  const int   g   = lane >> 4;
  const int   l15 = lane & 15;
  const float mu  = (float)l15 * (1.0f / 15.0f);  // linspace(0,1,16)
  const float* vinB = vin + (size_t)b * Nn * CIN;

  // ---- step A: per-point G = kernel_b^T (16xK) x vals (KxCIN) ----
  for (int p = 0; p < 16; ++p) {
    // lane needs r_k for k in [8g,8g+8) u [16+8g,16+8g+8): two contiguous runs
    union { float4 q[4]; float f[16]; } ra;
    ra.q[0] = *(const float4*)&rbuf[w][p][8 * g];
    ra.q[1] = *(const float4*)&rbuf[w][p][8 * g + 4];
    ra.q[2] = *(const float4*)&rbuf[w][p][16 + 8 * g];
    ra.q[3] = *(const float4*)&rbuf[w][p][16 + 8 * g + 4];
    // A fragment: A[e = l15, k], bf16 16x32 layout (k pairs per VGPR)
    BF16x16 afrag;
#pragma unroll
    for (int j = 0; j < 8; ++j) {
      int q0 = 2 * (j & 3) + 8 * (j >> 2);   // == local index of k0
      float d0 = ra.f[q0 + 0] - mu;
      float d1 = ra.f[q0 + 1] - mu;
      afrag.w[j] = f2bf2(__expf(-GAMMA * d0 * d0), __expf(-GAMMA * d1 * d1));
    }
    // lane needs idx for k in [8g,8g+8) u [16+8g,16+8g+8): two contiguous runs
    union { int4 q[4]; int s[16]; } ib;
    ib.q[0] = *(const int4*)&ibuf[w][p][8 * g];
    ib.q[1] = *(const int4*)&ibuf[w][p][8 * g + 4];
    ib.q[2] = *(const int4*)&ibuf[w][p][16 + 8 * g];
    ib.q[3] = *(const int4*)&ibuf[w][p][16 + 8 * g + 4];
    // B fragments: B[k, i = l15 + 16h] gathered from values_in (L2-hot)
    BF16x16 bf0, bf1;
#pragma unroll
    for (int j = 0; j < 8; ++j) {
      const float* rA = vinB + (size_t)ib.s[2 * j + 0] * CIN;
      const float* rB = vinB + (size_t)ib.s[2 * j + 1] * CIN;
      bf0.w[j] = f2bf2(rA[l15],      rB[l15]);
      bf1.w[j] = f2bf2(rA[l15 + 16], rB[l15 + 16]);
    }
    v8f acc0 = {}, acc1 = {};
    acc0 = __builtin_amdgcn_wmma_f32_16x16x32_bf16(false, afrag.v, false, bf0.v,
                                                   (short)0, acc0, false, false);
    acc1 = __builtin_amdgcn_wmma_f32_16x16x32_bf16(false, afrag.v, false, bf1.v,
                                                   (short)0, acc1, false, false);
    // spill G to LDS: lane owns c = i*16 + 8g + v (v=0..7 contiguous)
    union { uint4 q; unsigned int w4[4]; } pk;
    pk.w4[0] = f2bf2(acc0[0], acc0[1]);
    pk.w4[1] = f2bf2(acc0[2], acc0[3]);
    pk.w4[2] = f2bf2(acc0[4], acc0[5]);
    pk.w4[3] = f2bf2(acc0[6], acc0[7]);
    *(uint4*)&Gbuf[w][p][l15 * 16 + 8 * g] = pk.q;
    pk.w4[0] = f2bf2(acc1[0], acc1[1]);
    pk.w4[1] = f2bf2(acc1[2], acc1[3]);
    pk.w4[2] = f2bf2(acc1[4], acc1[5]);
    pk.w4[3] = f2bf2(acc1[6], acc1[7]);
    *(uint4*)&Gbuf[w][p][(l15 + 16) * 16 + 8 * g] = pk.q;
  }
  __syncthreads();

  // ---- step B: Out[16x32] = Gflat[16x512] x Cflat[512x32] ----
  v8f oacc0 = {}, oacc1 = {};
  const unsigned int* gRowU = (const unsigned int*)&Gbuf[w][l15][0];  // 256 u32
  const unsigned int* cRow0 = (const unsigned int*)(cflatT + (size_t)l15 * 512);
  const unsigned int* cRow1 = (const unsigned int*)(cflatT + (size_t)(l15 + 16) * 512);
  for (int t = 0; t < 16; ++t) {
    // A: ga.w[j] = gRowU[16t + 8g + j] -> two contiguous uint4
    BF16x16 ga, cb0, cb1;
    ga.q[0]  = *(const uint4*)&gRowU[16 * t + 8 * g];
    ga.q[1]  = *(const uint4*)&gRowU[16 * t + 8 * g + 4];
    // B: cbX.w[j] = cRowX[16t + 8g + j] -> two contiguous uint4 each
    cb0.q[0] = *(const uint4*)&cRow0[16 * t + 8 * g];
    cb0.q[1] = *(const uint4*)&cRow0[16 * t + 8 * g + 4];
    cb1.q[0] = *(const uint4*)&cRow1[16 * t + 8 * g];
    cb1.q[1] = *(const uint4*)&cRow1[16 * t + 8 * g + 4];
    oacc0 = __builtin_amdgcn_wmma_f32_16x16x32_bf16(false, ga.v, false, cb0.v,
                                                    (short)0, oacc0, false, false);
    oacc1 = __builtin_amdgcn_wmma_f32_16x16x32_bf16(false, ga.v, false, cb1.v,
                                                    (short)0, oacc1, false, false);
  }
  // epilogue: out = acc/K + bias
  const float b0 = sbias[l15];
  const float b1 = sbias[l15 + 16];
#pragma unroll
  for (int v = 0; v < 8; ++v) {
    int p = v + 8 * g;                       // C/D layout: M = v + 8*(lane>>4)
    size_t row = ((size_t)pBase + p) * COUT;
    out[row + l15]      = oacc0[v] * (1.0f / (float)Kk) + b0;
    out[row + l15 + 16] = oacc1[v] * (1.0f / (float)Kk) + b1;
  }
}

// ---------------------------------------------------------------------------
extern "C" void kernel_launch(void* const* d_in, const int* in_sizes, int n_in,
                              void* d_out, int out_size, void* d_ws, size_t ws_size,
                              hipStream_t stream) {
  const float* pin   = (const float*)d_in[0];  // points_in  [B,N,2]
  const float* vin   = (const float*)d_in[1];  // values_in  [B,N,CIN]
  const float* pout  = (const float*)d_in[2];  // points_out [B,M,2]
  const float* coeff = (const float*)d_in[3];  // coeff      [CIN,COUT,E]
  const float* bias  = (const float*)d_in[4];  // bias       [COUT]

  // d_out = uiv_k [B,M,K,2] ++ idx [B,M,K] (int bits) ++ out [B,M,COUT]
  float* outF = (float*)d_out;
  float* uiv  = outF;                                   // 1,048,576 f32
  int*   idxO = (int*)(outF + (size_t)Bc * Mm * Kk * 2);//   524,288 i32
  float* conv = outF + (size_t)Bc * Mm * Kk * 2 + (size_t)Bc * Mm * Kk;

  unsigned short* cflatT = (unsigned short*)d_ws;       // 32 KB

  prep_cflat<<<64, 256, 0, stream>>>(coeff, cflatT);
  knn_kernel<<<dim3(Mm / 256, Bc), 256, 0, stream>>>(pin, pout, uiv, idxO);
  conv_kernel<<<(Bc * Mm) / 32, 64, 0, stream>>>(vin, bias, cflatT, uiv, idxO, conv);
}
